// SelfAttention_76862734729316
// MI455X (gfx1250) — matrix operand order repaired and checked by reading, not compiled, not run on hardware
//
#include <hip/hip_runtime.h>
#include <hip/hip_bf16.h>
#include <math.h>

#define Bb 8
#define Cch 64
#define CQ 8
#define NN 4096

typedef __bf16 bf16;
typedef __attribute__((ext_vector_type(8)))  __bf16 bf16x8;
typedef __attribute__((ext_vector_type(16))) __bf16 bf16x16;
typedef __attribute__((ext_vector_type(8)))  float  f32x8;

// lane l <-> lane l^16 exchange, pure VALU (v_permlanex16), no DS stall.
__device__ __forceinline__ float xor16(float v) {
    int i = __builtin_bit_cast(int, v);
    int r = __builtin_amdgcn_permlanex16(i, i, 0x76543210, 0xfedcba98,
                                         false, false);
    return __builtin_bit_cast(float, r);
}

// ---------------------------------------------------------------------------
// Phase 1: 1x1-conv projections.  One thread per (b, n) spatial position.
//   f_t, g_t : [B][N][8]  bf16  (K-contiguous for WMMA A/B operand loads)
//   h_bf     : [B][64][N] bf16  (n-contiguous rows for value-matmul A operand)
// ---------------------------------------------------------------------------
__global__ __launch_bounds__(128) void proj_kernel(
    const float* __restrict__ x,
    const float* __restrict__ Wq, const float* __restrict__ bq,
    const float* __restrict__ Wk, const float* __restrict__ bk,
    const float* __restrict__ Wv, const float* __restrict__ bv,
    bf16* __restrict__ f_t, bf16* __restrict__ g_t, bf16* __restrict__ h_bf)
{
    const int b = blockIdx.y;
    const int n = blockIdx.x * 128 + threadIdx.x;

    float xv[Cch];
#pragma unroll
    for (int c = 0; c < Cch; ++c)
        xv[c] = x[((size_t)(b * Cch + c)) * NN + n];   // coalesced across lanes

    // f = Wq x + bq ; g = Wk x + bk   (8 outputs each)
    bf16x8 fo, go;
#pragma unroll
    for (int o = 0; o < CQ; ++o) {
        float aq = bq[o];
        float ak = bk[o];
#pragma unroll 16
        for (int c = 0; c < Cch; ++c) {
            aq = fmaf(Wq[o * Cch + c], xv[c], aq);
            ak = fmaf(Wk[o * Cch + c], xv[c], ak);
        }
        fo[o] = (bf16)aq;
        go[o] = (bf16)ak;
    }
    *(bf16x8*)(f_t + ((size_t)b * NN + n) * CQ) = fo;   // 16B, coalesced
    *(bf16x8*)(g_t + ((size_t)b * NN + n) * CQ) = go;

    // h = Wv x + bv   (64 outputs)
    for (int o = 0; o < Cch; ++o) {
        float av = bv[o];
#pragma unroll 16
        for (int c = 0; c < Cch; ++c)
            av = fmaf(Wv[o * Cch + c], xv[c], av);
        h_bf[((size_t)(b * Cch + o)) * NN + n] = (bf16)av;  // coalesced per o
    }
}

// ---------------------------------------------------------------------------
// Phase 2: flash-attention style streaming softmax-attention.
// grid = (N/64, B); 4 waves per block, each wave owns a 16-wide m-tile.
// Per 32-n chunk: all global loads issued first (overlap score WMMAs + exp
// chain), 2 score WMMAs (K=8 padded), online softmax with v_permlanex16
// reductions, P staged via wave-private LDS (in-order DS, no barrier needed)
// into B-operand layout, 4 value WMMAs (64 channels).
// ---------------------------------------------------------------------------
__global__ __launch_bounds__(128) void attn_kernel(
    const bf16* __restrict__ f_t, const bf16* __restrict__ g_t,
    const bf16* __restrict__ h_bf,
    const float* __restrict__ x, const float* __restrict__ gptr,
    float* __restrict__ out)
{
    const int b    = blockIdx.y;
    const int w    = threadIdx.x >> 5;     // wave id 0..3
    const int lane = threadIdx.x & 31;
    const int lr   = lane & 15;            // index within half-wave
    const int hi   = lane >> 4;            // 0 = lanes 0-15, 1 = lanes 16-31
    const int m    = blockIdx.x * 64 + w * 16 + lr;   // this lane's column

    // wave-private P staging tile: [m 0..15][k 0..31] bf16, pitch 40 (80B row)
    __shared__ __align__(16) bf16 plds[4][16][40];

    // --- B operand for score matmul: G tile (K=d 0..7, N=m), rest zero ---
    bf16x16 bg;
#pragma unroll
    for (int e = 0; e < 16; ++e) bg[e] = (bf16)0.0f;
    if (hi == 0) {
        bf16x8 gv = *(const bf16x8*)(g_t + ((size_t)b * NN + m) * CQ);
#pragma unroll
        for (int e = 0; e < 8; ++e) bg[e] = gv[e];   // lanes 0-15 hold K=0..7
    }

    // per-lane H row base pointers for the 4 channel tiles
    const bf16* hrow[4];
#pragma unroll
    for (int t = 0; t < 4; ++t)
        hrow[t] = h_bf + ((size_t)(b * Cch + t * 16 + lr)) * NN;
    const bf16* frow = f_t + ((size_t)b * NN) * CQ;

    f32x8 acc[4];
#pragma unroll
    for (int t = 0; t < 4; ++t)
#pragma unroll
        for (int r = 0; r < 8; ++r) acc[t][r] = 0.0f;

    float mrun = -INFINITY;
    float lrun = 0.0f;

    for (int n0 = 0; n0 < NN; n0 += 32) {
        // ================= issue ALL global loads up front =================
        // score A operands: F^T rows for chunks [n0,n0+16), [n0+16,n0+32)
        bf16x16 af0, af1;
#pragma unroll
        for (int e = 0; e < 16; ++e) { af0[e] = (bf16)0.0f; af1[e] = (bf16)0.0f; }
        if (hi == 0) {   // A layout: lanes 0-15 carry K=0..7 (V0-3); K>=8 zero
            bf16x8 v0 = *(const bf16x8*)(frow + (size_t)(n0 + lr) * CQ);
            bf16x8 v1 = *(const bf16x8*)(frow + (size_t)(n0 + 16 + lr) * CQ);
#pragma unroll
            for (int e = 0; e < 8; ++e) { af0[e] = v0[e]; af1[e] = v1[e]; }
        }
        // value A operands (consumed only after softmax -> latency hidden)
        bf16x8 h0[4], h1[4];
#pragma unroll
        for (int t = 0; t < 4; ++t) {
            h0[t] = *(const bf16x8*)(hrow[t] + n0 + 8 * hi);       // K 0-7 / 8-15
            h1[t] = *(const bf16x8*)(hrow[t] + n0 + 16 + 8 * hi);  // K 16-23 / 24-31
        }
        // prefetch next chunk (global_prefetch_b8)
        if (n0 + 32 < NN) {
#pragma unroll
            for (int t = 0; t < 4; ++t)
                __builtin_prefetch(hrow[t] + n0 + 32, 0, 0);
            __builtin_prefetch(frow + (size_t)(n0 + 32 + lr) * CQ, 0, 0);
        }

        // ========================= score WMMAs =============================
        f32x8 zc;
#pragma unroll
        for (int r = 0; r < 8; ++r) zc[r] = 0.0f;

        f32x8 s0 = __builtin_amdgcn_wmma_f32_16x16x32_bf16(
            false, af0, false, bg, (short)0, zc, false, false);
        f32x8 s1 = __builtin_amdgcn_wmma_f32_16x16x32_bf16(
            false, af1, false, bg, (short)0, zc, false, false);

        // ============ online softmax over n for this lane's column =========
        float cm = -INFINITY;
#pragma unroll
        for (int r = 0; r < 8; ++r) {
            cm = fmaxf(cm, s0[r]);
            cm = fmaxf(cm, s1[r]);
        }
        cm = fmaxf(cm, xor16(cm));          // combine lane pair (column split)
        const float mnew  = fmaxf(mrun, cm);
        const float scale = __expf(mrun - mnew);

        f32x8 p0, p1;
        float ls = 0.0f;
#pragma unroll
        for (int r = 0; r < 8; ++r) {
            p0[r] = __expf(s0[r] - mnew); ls += p0[r];
            p1[r] = __expf(s1[r] - mnew); ls += p1[r];
        }
        ls  += xor16(ls);
        lrun = lrun * scale + ls;
        mrun = mnew;
#pragma unroll
        for (int t = 0; t < 4; ++t) acc[t] = acc[t] * scale;

        // ===== stage P (bf16) into B-operand layout via wave-private LDS ===
        // same-wave DS ops are processed in order (ISA 7.3) -> no barrier.
        const int cb = hi * 8;   // this lane's D rows are n_local = r + 8*hi
#pragma unroll
        for (int r = 0; r < 8; ++r) {
            plds[w][lr][cb + r]      = (bf16)p0[r];
            plds[w][lr][16 + cb + r] = (bf16)p1[r];
        }
        bf16x16 bp;   // B 32x16: lanes 0-15 K=0..7,16..23, lanes 16-31 K=8..15,24..31
        {
            bf16x8 q0 = *(const bf16x8*)&plds[w][lr][hi * 8];
            bf16x8 q1 = *(const bf16x8*)&plds[w][lr][16 + hi * 8];
#pragma unroll
            for (int e = 0; e < 8; ++e) { bp[e] = q0[e]; bp[8 + e] = q1[e]; }
        }

        // ===== value matmul: acc[t] += H(16x32) x P(32x16), 4 tiles ========
#pragma unroll
        for (int t = 0; t < 4; ++t) {
            bf16x16 ah;
#pragma unroll
            for (int e = 0; e < 8; ++e) { ah[e] = h0[t][e]; ah[8 + e] = h1[t][e]; }
            acc[t] = __builtin_amdgcn_wmma_f32_16x16x32_bf16(
                false, ah, false, bp, (short)0, acc[t], false, false);
        }
    }

    // ---- epilogue: out = gamma * acc / l + x ----
    const float invl = 1.0f / lrun;
    const float gm   = gptr[0];
#pragma unroll
    for (int t = 0; t < 4; ++t) {
#pragma unroll
        for (int r = 0; r < 8; ++r) {
            const int c = t * 16 + r + 8 * hi;   // D row -> channel
            const size_t idx = ((size_t)(b * Cch + c)) * NN + m;
            out[idx] = gm * acc[t][r] * invl + x[idx];
        }
    }
}

// ---------------------------------------------------------------------------
extern "C" void kernel_launch(void* const* d_in, const int* in_sizes, int n_in,
                              void* d_out, int out_size, void* d_ws, size_t ws_size,
                              hipStream_t stream) {
    (void)in_sizes; (void)n_in; (void)out_size; (void)ws_size;
    const float* x     = (const float*)d_in[0];
    const float* Wq    = (const float*)d_in[1];
    const float* bq    = (const float*)d_in[2];
    const float* Wk    = (const float*)d_in[3];
    const float* bk    = (const float*)d_in[4];
    const float* Wv    = (const float*)d_in[5];
    const float* bv    = (const float*)d_in[6];
    const float* gamma = (const float*)d_in[7];
    float* out = (float*)d_out;

    // workspace: f_t [B][N][8] bf16, g_t [B][N][8] bf16, h_bf [B][64][N] bf16
    bf16* f_t  = (bf16*)d_ws;
    bf16* g_t  = f_t + (size_t)Bb * NN * CQ;
    bf16* h_bf = g_t + (size_t)Bb * NN * CQ;   // total ~5 MB

    proj_kernel<<<dim3(NN / 128, Bb), 128, 0, stream>>>(
        x, Wq, bq, Wk, bk, Wv, bv, f_t, g_t, h_bf);
    attn_kernel<<<dim3(NN / 64, Bb), 128, 0, stream>>>(
        f_t, g_t, h_bf, x, gamma, out);
}
